// DeformableAttention_15169824489901
// MI455X (gfx1250) — compile-verified, matrix operand order checked by reading, not run on hardware
//
#include <hip/hip_runtime.h>
#include <hip/hip_bf16.h>

// ---------------------------------------------------------------------------
// DeformableAttention for MI455X (gfx1250, wave32, WMMA).
// B=8, C=256, H=W=64, N=4096, NH=8, K=4, HD=32, BN=B*N=32768.
// GEMMs: bf16 WMMA (v_wmma_f32_16x16x32_bf16), f32 accumulate, double-buffered
// async global->LDS staging (ASYNCcnt) when the toolchain exposes the builtins.
// ---------------------------------------------------------------------------

typedef __bf16 bf16_t;
typedef __attribute__((ext_vector_type(16))) __bf16 v16bf;
typedef __attribute__((ext_vector_type(8)))  __bf16 v8bf;
typedef __attribute__((ext_vector_type(8)))  float  v8f;

// Async global->LDS builtin takes (v4i AS1*, v4i AS3*, imm offset, imm cpol);
// v4i is the GCC-vector int4 (exact pointee type required by clang).
typedef int v4i __attribute__((vector_size(16)));
typedef __attribute__((address_space(1))) v4i gv4i;
typedef __attribute__((address_space(3))) v4i lv4i;

#if __has_builtin(__builtin_amdgcn_global_load_async_to_lds_b128) && \
    __has_builtin(__builtin_amdgcn_s_wait_asynccnt)
#define DA_ASYNC_LDS 1
#else
#define DA_ASYNC_LDS 0
#endif

#define DA_B   8
#define DA_C   256
#define DA_H   64
#define DA_W   64
#define DA_N   4096          // H*W
#define DA_BN  32768         // B*N
#define DA_NH  8
#define DA_K   4
#define DA_HD  32
#define DA_NCAT 352          // C + NH*K*2 + NH*K = 256 + 64 + 32

// ------------------------------- WMMA helpers ------------------------------

// Load a 16x32 bf16 fragment (A layout; B uses the same pattern on W^T rows)
// from LDS. Rows are `ldr` bf16 elements apart. Matches CDNA5 ISA 16-bit
// A-matrix layout: lane<16 -> K {0..7, 16..23}, lane>=16 -> K {8..15, 24..31}.
__device__ __forceinline__ v16bf load_frag_bf16(const bf16_t* base, int ldr, int lane) {
  int row = lane & 15;
  const bf16_t* p = base + row * ldr + ((lane & 16) ? 8 : 0);
  v8bf lo = *(const v8bf*)(p);
  v8bf hi = *(const v8bf*)(p + 16);
  v16bf r;
#pragma unroll
  for (int i = 0; i < 8; ++i) { r[i] = lo[i]; r[i + 8] = hi[i]; }
  return r;
}

// Copy 64B global -> LDS (4 x b128). Async (ASYNCcnt) when available.
__device__ __forceinline__ void copy64B(bf16_t* dst, const bf16_t* src) {
#if DA_ASYNC_LDS
#pragma unroll
  for (int i = 0; i < 4; ++i)
    __builtin_amdgcn_global_load_async_to_lds_b128(
        (gv4i*)(src + i * 8), (lv4i*)(dst + i * 8), 0, 0);
#else
#pragma unroll
  for (int i = 0; i < 4; ++i)
    *(v8bf*)(dst + i * 8) = *(const v8bf*)(src + i * 8);
  __builtin_prefetch(src + 64, 0, 1);   // gfx1250 global_prefetch_b8
#endif
}

__device__ __forceinline__ void wait_stage() {
#if DA_ASYNC_LDS
  __builtin_amdgcn_s_wait_asynccnt(0);
#endif
  __syncthreads();
}

// ------------------------- Kernel 0: weight prep ---------------------------
// WcatT[n][k] = concat(Wv | Woff | Wa)^T in bf16  (352 x 256)
// WoutT[n][k] = Wout^T in bf16                    (256 x 256)
// biascat[n]  = concat(bv | boff | ba)            (352)
__global__ __launch_bounds__(256) void da_prep_weights(
    const float* __restrict__ Wv, const float* __restrict__ Woff,
    const float* __restrict__ Wa, const float* __restrict__ bv,
    const float* __restrict__ boff, const float* __restrict__ ba,
    const float* __restrict__ Wout,
    bf16_t* __restrict__ WcatT, bf16_t* __restrict__ WoutT,
    float* __restrict__ biascat) {
  int n = blockIdx.x;    // 0..351
  int k = threadIdx.x;   // 0..255
  float w;
  if (n < 256)      w = Wv[k * 256 + n];
  else if (n < 320) w = Woff[k * 64 + (n - 256)];
  else              w = Wa[k * 32 + (n - 320)];
  WcatT[n * 256 + k] = (bf16_t)w;
  if (n < 256) WoutT[n * 256 + k] = (bf16_t)Wout[k * 256 + n];
  if (k == 0)
    biascat[n] = (n < 256) ? bv[n] : (n < 320 ? boff[n - 256] : ba[n - 320]);
}

// --------------------- Kernel 1: transpose x to (B,N,C) --------------------
__global__ __launch_bounds__(256) void da_transpose_x(
    const float* __restrict__ x, float* __restrict__ xf, bf16_t* __restrict__ xb) {
  __shared__ float tile[32][33];
  int b  = blockIdx.z;
  int n0 = blockIdx.x * 32;
  int c0 = blockIdx.y * 32;
  int tx = threadIdx.x, ty = threadIdx.y;  // 32 x 8
#pragma unroll
  for (int i = 0; i < 4; ++i) {
    int cc = ty + 8 * i;
    tile[cc][tx] = x[((size_t)b * DA_C + c0 + cc) * DA_N + n0 + tx];
  }
  __syncthreads();
#pragma unroll
  for (int i = 0; i < 4; ++i) {
    int nn = ty + 8 * i;
    float v = tile[tx][nn];
    size_t idx = ((size_t)b * DA_N + n0 + nn) * DA_C + c0 + tx;
    xf[idx] = v;
    xb[idx] = (bf16_t)v;
  }
}

// ------------------------ Kernels 2 & 4: bf16 WMMA GEMM --------------------
// C[M x Nn] = A[M x K] * BT[Nn x K]^T + bias (+ resid)
// 256 threads = 8 waves. BM=128, BN=32, BK=64. Each wave owns one 16-row
// slice and two 16-col tiles (A-fragment reused across 2 WMMAs); B tile is
// shared by all 8 waves. Double-buffered async global->LDS staging: one
// barrier per K=64 chunk.
__global__ __launch_bounds__(256) void da_gemm_bf16(
    const bf16_t* __restrict__ A, const bf16_t* __restrict__ BT,
    const float* __restrict__ bias, const float* __restrict__ resid,
    float* __restrict__ Cf, int M, int Nn, int Kk) {
  constexpr int BM = 128, BN = 32, BK = 64;
  constexpr int LDR = BK + 8;  // 72-elem (144B) pitch: conflict-free ds_load_b128
  __shared__ __align__(16) bf16_t Als[2][BM * LDR];
  __shared__ __align__(16) bf16_t Bls[2][BN * LDR];

  const int tid  = threadIdx.x;
  const int wave = tid >> 5;
  const int lane = tid & 31;
  const int m0 = blockIdx.x * BM;
  const int n0 = blockIdx.y * BN;

  // Stage one K-chunk (A: 128x64, B: 32x64) into LDS buffer `buf`.
  auto stage = [&](int k0, int buf) {
    {
      int r = tid >> 1, seg = (tid & 1) * 32;           // 2 threads per 128B row
      copy64B(&Als[buf][r * LDR + seg],
              A + (size_t)(m0 + r) * Kk + k0 + seg);
    }
    if (tid < 64) {
      int r = tid >> 1, seg = (tid & 1) * 32;
      copy64B(&Bls[buf][r * LDR + seg],
              BT + (size_t)(n0 + r) * Kk + k0 + seg);
    }
  };

  v8f acc0 = {}, acc1 = {};
  const int nchunks = Kk / BK;

  stage(0, 0);
  for (int c = 0; c < nchunks; ++c) {
    const int buf = c & 1;
    wait_stage();                         // own async copies landed + all waves past
    if (c + 1 < nchunks) stage((c + 1) * BK, buf ^ 1);
#pragma unroll
    for (int s = 0; s < 2; ++s) {         // two K=32 sub-chunks
      v16bf af = load_frag_bf16(&Als[buf][(wave * 16) * LDR + s * 32], LDR, lane);
      v16bf b0 = load_frag_bf16(&Bls[buf][0 * LDR + s * 32], LDR, lane);
      v16bf b1 = load_frag_bf16(&Bls[buf][16 * LDR + s * 32], LDR, lane);
      acc0 = __builtin_amdgcn_wmma_f32_16x16x32_bf16(
          false, af, false, b0, (short)0, acc0, false, false);
      acc1 = __builtin_amdgcn_wmma_f32_16x16x32_bf16(
          false, af, false, b1, (short)0, acc1, false, false);
    }
    // no trailing barrier needed: next iteration's wait_stage() barrier
    // separates these LDS reads from the restaging of this buffer.
  }

  // D layout: lane&15 = column, lanes>=16 hold rows 8..15, VGPR j = row j(+8)
  const int col0  = n0 + (lane & 15);
  const int col1  = col0 + 16;
  const int mbase = m0 + wave * 16 + ((lane >> 4) << 3);
  const float bb0 = bias ? bias[col0] : 0.0f;
  const float bb1 = bias ? bias[col1] : 0.0f;
#pragma unroll
  for (int j = 0; j < 8; ++j) {
    size_t i0 = (size_t)(mbase + j) * Nn + col0;
    size_t i1 = (size_t)(mbase + j) * Nn + col1;
    float v0 = acc0[j] + bb0;
    float v1 = acc1[j] + bb1;
    if (resid) { v0 += resid[i0]; v1 += resid[i1]; }
    Cf[i0] = v0;
    Cf[i1] = v1;
  }
}

// ----------------- Kernel 3: softmax + bilinear sampling -------------------
// One wave per (b,n,nh); lane = head-dim channel. Each corner gather reads
// 32 contiguous floats (128B) from the v slice of `cat`.
__global__ __launch_bounds__(256) void da_sample(
    const float* __restrict__ cat,       // BN x 352
    bf16_t* __restrict__ out_att) {      // BN x 256 (bf16 A for GEMM2)
  int tid  = threadIdx.x;
  int wave = tid >> 5;
  int lane = tid & 31;
  long task = (long)blockIdx.x * 8 + wave;   // 0 .. B*N*NH-1
  int  nh = (int)(task & (DA_NH - 1));
  long bn = task >> 3;
  int  n  = (int)(bn & (DA_N - 1));
  int  b  = (int)(bn >> 12);

  const float* row = cat + bn * DA_NCAT;

  float lg[DA_K], ox[DA_K], oy[DA_K];
  float mx = -1e30f;
#pragma unroll
  for (int kk = 0; kk < DA_K; ++kk) {
    lg[kk] = row[320 + nh * DA_K + kk];
    mx = fmaxf(mx, lg[kk]);
    ox[kk] = row[256 + (nh * DA_K + kk) * 2 + 0];
    oy[kk] = row[256 + (nh * DA_K + kk) * 2 + 1];
  }
  float se = 0.f;
#pragma unroll
  for (int kk = 0; kk < DA_K; ++kk) { lg[kk] = __expf(lg[kk] - mx); se += lg[kk]; }
  float inv = 1.0f / se;

  int px = n & (DA_W - 1), py = n >> 6;
  float gx = -1.0f + 2.0f * (float)px * (1.0f / (DA_W - 1));
  float gy = -1.0f + 2.0f * (float)py * (1.0f / (DA_H - 1));
  const float scale = 2.0f / 64.0f;

  const float* vb = cat + ((long)b * DA_N) * DA_NCAT + nh * DA_HD + lane;
  float acc = 0.f;
#pragma unroll
  for (int kk = 0; kk < DA_K; ++kk) {
    float cx = fminf(fmaxf(gx + ox[kk] * scale, -1.f), 1.f);
    float cy = fminf(fmaxf(gy + oy[kk] * scale, -1.f), 1.f);
    float ix = (cx + 1.f) * 0.5f * (float)(DA_W - 1);
    float iy = (cy + 1.f) * 0.5f * (float)(DA_H - 1);
    float fx = floorf(ix), fy = floorf(iy);
    float wx1 = ix - fx, wy1 = iy - fy;
    float wx0 = 1.f - wx1, wy0 = 1.f - wy1;
    int x0 = (int)fx;     x0 = x0 < 0 ? 0 : (x0 > DA_W - 1 ? DA_W - 1 : x0);
    int x1 = (int)fx + 1; x1 = x1 < 0 ? 0 : (x1 > DA_W - 1 ? DA_W - 1 : x1);
    int y0 = (int)fy;     y0 = y0 < 0 ? 0 : (y0 > DA_H - 1 ? DA_H - 1 : y0);
    int y1 = (int)fy + 1; y1 = y1 < 0 ? 0 : (y1 > DA_H - 1 ? DA_H - 1 : y1);
    float v00 = vb[(long)(y0 * DA_W + x0) * DA_NCAT];
    float v10 = vb[(long)(y0 * DA_W + x1) * DA_NCAT];
    float v01 = vb[(long)(y1 * DA_W + x0) * DA_NCAT];
    float v11 = vb[(long)(y1 * DA_W + x1) * DA_NCAT];
    float s = v00 * (wx0 * wy0) + v10 * (wx1 * wy0)
            + v01 * (wx0 * wy1) + v11 * (wx1 * wy1);
    acc += lg[kk] * inv * s;
  }
  out_att[bn * DA_C + nh * DA_HD + lane] = (bf16_t)acc;
}

// --------------------- Kernel 5: layernorm statistics ----------------------
__global__ __launch_bounds__(256) void da_stats(
    const float* __restrict__ y, float* __restrict__ mu, float* __restrict__ rs) {
  int tid = threadIdx.x;
  int wave = tid >> 5, lane = tid & 31;
  long rowid = (long)blockIdx.x * 8 + wave;
  const float* row = y + rowid * DA_C;
  float s = 0.f, s2 = 0.f;
#pragma unroll
  for (int i = 0; i < 8; ++i) {
    float v = row[lane + 32 * i];
    s += v; s2 += v * v;
  }
#pragma unroll
  for (int d = 16; d > 0; d >>= 1) {
    s  += __shfl_xor(s,  d, 32);
    s2 += __shfl_xor(s2, d, 32);
  }
  if (lane == 0) {
    float m   = s  * (1.0f / DA_C);
    float var = s2 * (1.0f / DA_C) - m * m;
    mu[rowid] = m;
    rs[rowid] = rsqrtf(var + 1e-5f);
  }
}

// --------------- Kernel 6: normalize + transpose to (B,C,H,W) --------------
__global__ __launch_bounds__(256) void da_finalize(
    const float* __restrict__ y, const float* __restrict__ mu,
    const float* __restrict__ rs, const float* __restrict__ gamma,
    const float* __restrict__ beta, float* __restrict__ out) {
  __shared__ float tile[32][33];
  int b  = blockIdx.z;
  int n0 = blockIdx.x * 32;
  int c0 = blockIdx.y * 32;
  int tx = threadIdx.x, ty = threadIdx.y;  // 32 x 8
#pragma unroll
  for (int i = 0; i < 4; ++i) {
    int nn = ty + 8 * i;
    long rowid = (long)b * DA_N + n0 + nn;
    float v = y[rowid * DA_C + c0 + tx];
    v = (v - mu[rowid]) * rs[rowid] * gamma[c0 + tx] + beta[c0 + tx];
    tile[nn][tx] = v;   // tile[n-in-tile][c-in-tile]
  }
  __syncthreads();
#pragma unroll
  for (int i = 0; i < 4; ++i) {
    int cc = ty + 8 * i;
    out[((long)b * DA_C + c0 + cc) * DA_N + n0 + tx] = tile[tx][cc];
  }
}

// ------------------------------- launcher ----------------------------------
static inline size_t align256(size_t v) { return (v + 255) & ~(size_t)255; }

extern "C" void kernel_launch(void* const* d_in, const int* in_sizes, int n_in,
                              void* d_out, int out_size, void* d_ws, size_t ws_size,
                              hipStream_t stream) {
  const float* x    = (const float*)d_in[0];
  const float* Wv   = (const float*)d_in[1];
  const float* bv   = (const float*)d_in[2];
  const float* Woff = (const float*)d_in[3];
  const float* boff = (const float*)d_in[4];
  const float* Wa   = (const float*)d_in[5];
  const float* ba   = (const float*)d_in[6];
  const float* Wout = (const float*)d_in[7];
  const float* bout = (const float*)d_in[8];
  const float* gamma= (const float*)d_in[9];
  const float* beta = (const float*)d_in[10];
  float* out = (float*)d_out;

  // Workspace carve-out (~97 MB). xb is reused as out_att, cat is reused as y.
  char* ws = (char*)d_ws;
  size_t off = 0;
  float*  xf      = (float*)(ws + off); off = align256(off + (size_t)DA_BN * DA_C * 4);
  bf16_t* xb      = (bf16_t*)(ws + off);                       // aliased: out_att
  size_t  xb_off  = off;                off = align256(off + (size_t)DA_BN * DA_C * 2);
  float*  cat     = (float*)(ws + off);                        // aliased: y
  size_t  cat_off = off;                off = align256(off + (size_t)DA_BN * DA_NCAT * 4);
  bf16_t* WcatT   = (bf16_t*)(ws + off); off = align256(off + (size_t)DA_NCAT * DA_C * 2);
  bf16_t* WoutT   = (bf16_t*)(ws + off); off = align256(off + (size_t)DA_C * DA_C * 2);
  float*  biascat = (float*)(ws + off);  off = align256(off + (size_t)DA_NCAT * 4);
  float*  mu      = (float*)(ws + off);  off = align256(off + (size_t)DA_BN * 4);
  float*  rs      = (float*)(ws + off);  off = align256(off + (size_t)DA_BN * 4);
  bf16_t* out_att = (bf16_t*)(ws + xb_off);
  float*  y       = (float*)(ws + cat_off);
  (void)ws_size; (void)in_sizes; (void)n_in; (void)out_size;

  // 0) weight prep
  da_prep_weights<<<dim3(DA_NCAT), dim3(256), 0, stream>>>(
      Wv, Woff, Wa, bv, boff, ba, Wout, WcatT, WoutT, biascat);

  // 1) x (B,C,N) -> x_flat (B,N,C) f32 + bf16
  da_transpose_x<<<dim3(DA_N / 32, DA_C / 32, DA_B), dim3(32, 8), 0, stream>>>(
      x, xf, xb);

  // 2) fused GEMM: cat = x_flat @ [Wv|Woff|Wa] + bias   (32768 x 352)
  da_gemm_bf16<<<dim3(DA_BN / 128, DA_NCAT / 32), dim3(256), 0, stream>>>(
      xb, WcatT, biascat, nullptr, cat, DA_BN, DA_NCAT, DA_C);

  // 3) softmax + bilinear sampling -> out_att (bf16)
  da_sample<<<dim3((DA_B * DA_N * DA_NH) / 8), dim3(256), 0, stream>>>(
      cat, out_att);

  // 4) GEMM: y = out_att @ Wout + bout + x_flat (residual)
  da_gemm_bf16<<<dim3(DA_BN / 128, DA_C / 32), dim3(256), 0, stream>>>(
      out_att, WoutT, bout, xf, y, DA_BN, DA_C, DA_C);

  // 5) layernorm stats per row
  da_stats<<<dim3(DA_BN / 8), dim3(256), 0, stream>>>(y, mu, rs);

  // 6) normalize + transpose to (B,C,H,W)
  da_finalize<<<dim3(DA_N / 32, DA_C / 32, DA_B), dim3(32, 8), 0, stream>>>(
      y, mu, rs, gamma, beta, out);
}
